// Dipole_36756330119247
// MI455X (gfx1250) — matrix operand order repaired and checked by reading, not compiled, not run on hardware
//
#include <hip/hip_runtime.h>
#include <math.h>

// ---------------------------------------------------------------------------
// Dipole model on MI455X (gfx1250, wave32, WMMA).
// All GEMMs use V_WMMA_F32_16X16X32_BF16 (bf16 inputs, fp32 accumulate).
// Round-3 change vs round-2 (driven by disasm): the fused recurrent-step
// kernel was register-bound (6 B-frags x 2 buffers), collapsing the double
// buffer into just-in-time loads with s_wait_loadcnt 0 before every WMMA.
// Now 12 waves x 4 N-tiles per wave (384 threads): ~130 VGPRs, live double
// buffer, and 50% more waves per block to hide latency on the sequential
// recurrence that dominates this workload.
// ---------------------------------------------------------------------------

typedef __bf16 bf16;
typedef __attribute__((ext_vector_type(16))) __bf16 v16bf_t;
typedef __attribute__((ext_vector_type(8)))  __bf16 v8bf_t;
typedef __attribute__((ext_vector_type(8)))  float  v8f_t;

#define T_DIM 64
#define B_DIM 32
#define D_IN  4096
#define D_DAY 256
#define H_DIM 256
#define D_OUTN 942
#define D_OUTP 1024             /* padded rows for W_o (grid coverage) */
#define ROWS  (T_DIM * B_DIM)   /* 2048 */
#define G3H   (3 * H_DIM)       /* 768  */
#define REV_BLOCKS (ROWS / 16)  /* 128  */
#define STEP_WAVES 12           /* waves per gru_step block */

__device__ __forceinline__ float sigf(float x) { return 1.0f / (1.0f + __expf(-x)); }

__device__ __forceinline__ v8f_t wmma_bf16(v16bf_t a, v16bf_t b, v8f_t c) {
  // D = A(16x32) x B(32x16) + C, fp32 accumulate
  return __builtin_amdgcn_wmma_f32_16x16x32_bf16(false, a, false, b, (short)0, c, false, false);
}

union AF { v16bf_t v; v8bf_t h[2]; };

// --------------------------- generic bf16 GEMM ------------------------------
// C[M,N] = A[M,K] * W[N,K]^T + bias ; ACT: 0=none, 1=sigmoid.
// Block: 256 threads = 8 waves (2x4). Wave tile 32x64 (2x4 WMMA tiles).
// Block tile 64x256. REQUIREMENTS (caller-guaranteed): M multiple of 64;
// W allocated with >= gridDim.x*256 rows (pad columns are discarded by the
// n<N store guard, so pad contents are irrelevant). K multiple of 32,
// row strides 32B-aligned. A-frag: two 16B loads/lane; B-frag: 32B load/lane.
template <int ACT, bool OUTF, bool OUTB>
__global__ __launch_bounds__(256) void gemm_bf16_k(
    const bf16* __restrict__ A, const bf16* __restrict__ W,
    const float* __restrict__ bias,
    float* __restrict__ Cf, bf16* __restrict__ Cb,
    int M, int N, int K) {
  const int lane = threadIdx.x & 31;
  const int wave = threadIdx.x >> 5;
  const int wm = wave >> 2, wn = wave & 3;
  const int m0 = blockIdx.y * 64 + wm * 32;
  const int n0 = blockIdx.x * 256 + wn * 64;
  const int lm = lane & 15, lg = lane >> 4;

  v8f_t acc[2][4];
#pragma unroll
  for (int mi = 0; mi < 2; ++mi)
#pragma unroll
    for (int ni = 0; ni < 4; ++ni)
#pragma unroll
      for (int r = 0; r < 8; ++r) acc[mi][ni][r] = 0.0f;

  // fold the lane sub-offsets into the base pointers
  const bf16* Arow0 = A + (size_t)(m0 + lm) * K + lg * 16;
  const bf16* Arow1 = A + (size_t)(m0 + 16 + lm) * K + lg * 16;
  const bf16* Wrow[4];
#pragma unroll
  for (int ni = 0; ni < 4; ++ni)
    Wrow[ni] = W + (size_t)(n0 + ni * 16 + lm) * K + lg * 16;

  // ---- double-buffered K loop: prefetch kk+32 while doing WMMA on kk ----
  AF a0c, a1c;
  v16bf_t bc[4];
  a0c.h[0] = *(const v8bf_t*)(Arow0);
  a0c.h[1] = *(const v8bf_t*)(Arow0 + 8);
  a1c.h[0] = *(const v8bf_t*)(Arow1);
  a1c.h[1] = *(const v8bf_t*)(Arow1 + 8);
#pragma unroll
  for (int ni = 0; ni < 4; ++ni) bc[ni] = *(const v16bf_t*)(Wrow[ni]);

  for (int kk = 32; kk < K; kk += 32) {
    AF a0n, a1n;
    v16bf_t bn[4];
    a0n.h[0] = *(const v8bf_t*)(Arow0 + kk);
    a0n.h[1] = *(const v8bf_t*)(Arow0 + kk + 8);
    a1n.h[0] = *(const v8bf_t*)(Arow1 + kk);
    a1n.h[1] = *(const v8bf_t*)(Arow1 + kk + 8);
#pragma unroll
    for (int ni = 0; ni < 4; ++ni) bn[ni] = *(const v16bf_t*)(Wrow[ni] + kk);
#pragma unroll
    for (int ni = 0; ni < 4; ++ni) {
      acc[0][ni] = wmma_bf16(a0c.v, bc[ni], acc[0][ni]);
      acc[1][ni] = wmma_bf16(a1c.v, bc[ni], acc[1][ni]);
    }
    a0c = a0n; a1c = a1n;
#pragma unroll
    for (int ni = 0; ni < 4; ++ni) bc[ni] = bn[ni];
  }
#pragma unroll
  for (int ni = 0; ni < 4; ++ni) {
    acc[0][ni] = wmma_bf16(a0c.v, bc[ni], acc[0][ni]);
    acc[1][ni] = wmma_bf16(a1c.v, bc[ni], acc[1][ni]);
  }

#pragma unroll
  for (int mi = 0; mi < 2; ++mi) {
#pragma unroll
    for (int ni = 0; ni < 4; ++ni) {
      const int n = n0 + ni * 16 + lm;
      if (n >= N) continue;
      const float bv = bias[n];
#pragma unroll
      for (int r = 0; r < 8; ++r) {
        const int m = m0 + mi * 16 + lg * 8 + r;
        float v = acc[mi][ni][r] + bv;
        if (ACT == 1) v = sigf(v);
        if (OUTF) Cf[(size_t)m * N + n] = v;
        if (OUTB) Cb[(size_t)m * N + n] = (bf16)v;
      }
    }
  }
}

// --------------- fused GRU step: rev matrix + fwd in one launch -------------
// Blocks [0, 128): reverse-GRU rows (i*32+b), blocks [128, 130): forward rows.
// Each block (384 threads = 12 waves): gh = h_prev @ Whh^T + bhh via WMMA
// (16 rows x 768 gates; each wave owns 4 N-tiles = 64 gate columns) into LDS,
// then the GRU cell elementwise with in-place state update.
__global__ __launch_bounds__(384) void gru_step_k(
    const bf16* __restrict__ Whh_r, const float* __restrict__ bhh_r,
    const float* __restrict__ gi_r, float* __restrict__ hr,
    bf16* __restrict__ hr_b, bf16* __restrict__ rev_bf,
    float* __restrict__ scores_rev, const float* __restrict__ attn_w,
    const bf16* __restrict__ Whh_f, const float* __restrict__ bhh_f,
    const float* __restrict__ gi_f, float* __restrict__ hf,
    bf16* __restrict__ hf_b, float* __restrict__ fwd_out,
    int step) {
  __shared__ float gh_s[16 * G3H];  // 48 KB of the 320 KB WGP LDS
  const bool isFwd = blockIdx.x >= REV_BLOCKS;
  const int r0 = (isFwd ? (blockIdx.x - REV_BLOCKS) : blockIdx.x) * 16;
  const bf16* Whh = isFwd ? Whh_f : Whh_r;
  const float* bhh = isFwd ? bhh_f : bhh_r;
  const float* gi = isFwd ? gi_f : gi_r;
  float* hstate = isFwd ? hf : hr;
  bf16* hstate_bf = isFwd ? hf_b : hr_b;

  const int lane = threadIdx.x & 31;
  const int wave = threadIdx.x >> 5;  // 0..11
  const int n0 = wave * 64;           // each wave: 4 N-tiles of 16
  const int lm = lane & 15, lg = lane >> 4;

  v8f_t acc[4];
#pragma unroll
  for (int ni = 0; ni < 4; ++ni)
#pragma unroll
    for (int r = 0; r < 8; ++r) acc[ni][r] = 0.0f;

  const bf16* Arow = hstate_bf + (size_t)(r0 + lm) * H_DIM + lg * 16;
  const bf16* Wrow[4];
#pragma unroll
  for (int ni = 0; ni < 4; ++ni)
    Wrow[ni] = Whh + (size_t)(n0 + ni * 16 + lm) * H_DIM + lg * 16;

  AF ac;
  v16bf_t bc[4];
  ac.h[0] = *(const v8bf_t*)(Arow);
  ac.h[1] = *(const v8bf_t*)(Arow + 8);
#pragma unroll
  for (int ni = 0; ni < 4; ++ni) bc[ni] = *(const v16bf_t*)(Wrow[ni]);

#pragma unroll
  for (int kk = 32; kk < H_DIM; kk += 32) {
    AF an;
    v16bf_t bn[4];
    an.h[0] = *(const v8bf_t*)(Arow + kk);
    an.h[1] = *(const v8bf_t*)(Arow + kk + 8);
#pragma unroll
    for (int ni = 0; ni < 4; ++ni) bn[ni] = *(const v16bf_t*)(Wrow[ni] + kk);
#pragma unroll
    for (int ni = 0; ni < 4; ++ni) acc[ni] = wmma_bf16(ac.v, bc[ni], acc[ni]);
    ac = an;
#pragma unroll
    for (int ni = 0; ni < 4; ++ni) bc[ni] = bn[ni];
  }
#pragma unroll
  for (int ni = 0; ni < 4; ++ni) acc[ni] = wmma_bf16(ac.v, bc[ni], acc[ni]);

#pragma unroll
  for (int ni = 0; ni < 4; ++ni) {
    const int n = n0 + ni * 16 + lm;
    const float bv = bhh[n];
#pragma unroll
    for (int r = 0; r < 8; ++r) gh_s[(lg * 8 + r) * G3H + n] = acc[ni][r] + bv;
  }
  __syncthreads();

  // GRU cell: one wave per row (waves 0..3 take a second row), 8 elems/lane
  for (int rl = wave; rl < 16; rl += STEP_WAVES) {
    const int grow = r0 + rl;
    int gi_row;
    if (isFwd) {
      gi_row = step * B_DIM + grow;  // grow == b
    } else {
      const int i = grow >> 5, b = grow & 31;
      int tsrc = i - step;
      if (tsrc < 0) tsrc = 0;        // clipped reverse gather
      gi_row = tsrc * B_DIM + b;
    }
    const float* gib = gi + (size_t)gi_row * G3H;
    const float* ghb = gh_s + rl * G3H;
    float* hrow = hstate + (size_t)grow * H_DIM;
    bf16* hbrow = hstate_bf + (size_t)grow * H_DIM;
    float ssum = 0.0f;
#pragma unroll
    for (int e = 0; e < 8; ++e) {
      const int h = lane + 32 * e;
      const float r = sigf(gib[h] + ghb[h]);
      const float z = sigf(gib[H_DIM + h] + ghb[H_DIM + h]);
      const float nn = tanhf(gib[2 * H_DIM + h] + r * ghb[2 * H_DIM + h]);
      const float hn = (1.0f - z) * nn + z * hrow[h];
      hrow[h] = hn;
      hbrow[h] = (bf16)hn;
      if (isFwd) {
        fwd_out[((size_t)step * B_DIM + grow) * H_DIM + h] = hn;
      } else {
        const int i = grow >> 5, b = grow & 31;
        rev_bf[(((size_t)i * T_DIM + step) * B_DIM + b) * H_DIM + h] = (bf16)hn;
        ssum += hn * attn_w[H_DIM + h];
      }
    }
    if (!isFwd) {
#pragma unroll
      for (int o = 16; o > 0; o >>= 1) ssum += __shfl_xor(ssum, o, 32);
      if (lane == 0) {
        const int i = grow >> 5, b = grow & 31;
        scores_rev[((size_t)i * T_DIM + step) * B_DIM + b] = ssum;
      }
    }
  }
}

// ------------------------- small elementwise kernels ------------------------
__global__ void cvt_f2b_k(const float* __restrict__ s, bf16* __restrict__ d, int n) {
  const int i = blockIdx.x * 256 + threadIdx.x;
  if (i < n) d[i] = (bf16)s[i];
}
__global__ void zero_f_k(float* __restrict__ d, int n) {
  const int i = blockIdx.x * 256 + threadIdx.x;
  if (i < n) d[i] = 0.0f;
}
__global__ void zero_b_k(bf16* __restrict__ d, int n) {
  const int i = blockIdx.x * 256 + threadIdx.x;
  if (i < n) d[i] = (bf16)0.0f;
}

__global__ __launch_bounds__(256) void fwd_score_k(const float* __restrict__ fwd_out,
                                                   const float* __restrict__ attn_w,
                                                   float* __restrict__ fscores) {
  const int row = blockIdx.x * 8 + (threadIdx.x >> 5);  // t*32+b
  const int lane = threadIdx.x & 31;
  const float* f = fwd_out + (size_t)row * H_DIM;
  float s = 0.0f;
#pragma unroll
  for (int e = 0; e < 8; ++e) s += f[lane + 32 * e] * attn_w[lane + 32 * e];
#pragma unroll
  for (int o = 16; o > 0; o >>= 1) s += __shfl_xor(s, o, 32);
  if (lane == 0) fscores[row] = s;
}

// Masked softmax over t<=i, then c_fwd/c_rev contexts and the 4H concat row.
__global__ __launch_bounds__(256) void attn_combine_k(
    const float* __restrict__ fscores, const float* __restrict__ scores_rev,
    const float* __restrict__ attn_b, const float* __restrict__ fwd_out,
    const bf16* __restrict__ rev_bf, bf16* __restrict__ ht_bf) {
  const int row = blockIdx.x;  // i*32 + b
  const int i = row >> 5, b = row & 31;
  const int tid = threadIdx.x;
  __shared__ float alpha[T_DIM];
  __shared__ float stats[2];
  if (tid < T_DIM) {
    alpha[tid] = (tid <= i)
        ? fscores[tid * B_DIM + b] +
              scores_rev[((size_t)i * T_DIM + tid) * B_DIM + b] + attn_b[0]
        : 0.0f;
  }
  __syncthreads();
  if (tid == 0) {
    float mx = alpha[0];
    for (int t = 1; t <= i; ++t) mx = fmaxf(mx, alpha[t]);
    float sm = 0.0f;
    for (int t = 0; t <= i; ++t) sm += __expf(alpha[t] - mx);
    stats[0] = mx;
    stats[1] = 1.0f / sm;
  }
  __syncthreads();
  if (tid < T_DIM) alpha[tid] = (tid <= i) ? __expf(alpha[tid] - stats[0]) * stats[1] : 0.0f;
  __syncthreads();
  const int h = tid;  // 256 threads == H_DIM
  float cf = 0.0f, cr = 0.0f;
  for (int t = 0; t <= i; ++t) {
    const float a = alpha[t];
    cf += a * fwd_out[((size_t)t * B_DIM + b) * H_DIM + h];
    cr += a * (float)rev_bf[(((size_t)i * T_DIM + t) * B_DIM + b) * H_DIM + h];
  }
  const float inv = 1.0f / (float)(i + 1);
  bf16* o = ht_bf + (size_t)row * (4 * H_DIM);
  o[h] = (bf16)(cf * inv);
  o[H_DIM + h] = (bf16)(cr * inv);
  o[2 * H_DIM + h] = (bf16)fwd_out[((size_t)i * B_DIM + b) * H_DIM + h];
  o[3 * H_DIM + h] = rev_bf[(((size_t)i * T_DIM + i) * B_DIM + b) * H_DIM + h];
}

// ------------------------------- launcher -----------------------------------
extern "C" void kernel_launch(void* const* d_in, const int* in_sizes, int n_in,
                              void* d_out, int out_size, void* d_ws, size_t ws_size,
                              hipStream_t stream) {
  const float* x     = (const float*)d_in[0];
  const float* Wemb  = (const float*)d_in[1];
  const float* bemb  = (const float*)d_in[2];
  const float* Wihf  = (const float*)d_in[3];
  const float* Whhf  = (const float*)d_in[4];
  const float* bihf  = (const float*)d_in[5];
  const float* bhhf  = (const float*)d_in[6];
  const float* Wihr  = (const float*)d_in[7];
  const float* Whhr  = (const float*)d_in[8];
  const float* bihr  = (const float*)d_in[9];
  const float* bhhr  = (const float*)d_in[10];
  const float* attnw = (const float*)d_in[11];
  const float* attnb = (const float*)d_in[12];
  const float* Wao   = (const float*)d_in[13];
  const float* bao   = (const float*)d_in[14];
  const float* Wo    = (const float*)d_in[15];
  const float* bo    = (const float*)d_in[16];
  float* out = (float*)d_out;
  (void)in_sizes; (void)n_in; (void)out_size; (void)ws_size;

  // bump allocator over d_ws (~113 MB; rev tensor stored bf16 = 64 MB,
  // which fits in the 192 MB L2 between producer and consumer).
  char* p = (char*)d_ws;
  auto alloc = [&](size_t bytes) -> void* {
    void* r = (void*)p;
    p += (bytes + 255) & ~(size_t)255;
    return r;
  };
  bf16*  xbf    = (bf16*)alloc((size_t)ROWS * D_IN * sizeof(bf16));
  bf16*  Wemb_b = (bf16*)alloc((size_t)D_DAY * D_IN * sizeof(bf16));
  bf16*  Wihf_b = (bf16*)alloc((size_t)G3H * D_DAY * sizeof(bf16));
  bf16*  Whhf_b = (bf16*)alloc((size_t)G3H * H_DIM * sizeof(bf16));
  bf16*  Wihr_b = (bf16*)alloc((size_t)G3H * D_DAY * sizeof(bf16));
  bf16*  Whhr_b = (bf16*)alloc((size_t)G3H * H_DIM * sizeof(bf16));
  bf16*  Wao_b  = (bf16*)alloc((size_t)D_DAY * 4 * H_DIM * sizeof(bf16));
  bf16*  Wo_b   = (bf16*)alloc((size_t)D_OUTP * D_DAY * sizeof(bf16));  // padded to 1024 rows
  bf16*  demb_b = (bf16*)alloc((size_t)ROWS * D_DAY * sizeof(bf16));
  float* gi_f   = (float*)alloc((size_t)ROWS * G3H * sizeof(float));
  float* gi_r   = (float*)alloc((size_t)ROWS * G3H * sizeof(float));
  float* hf     = (float*)alloc((size_t)B_DIM * H_DIM * sizeof(float));
  bf16*  hf_b   = (bf16*)alloc((size_t)B_DIM * H_DIM * sizeof(bf16));
  float* fwd_o  = (float*)alloc((size_t)ROWS * H_DIM * sizeof(float));
  float* hr     = (float*)alloc((size_t)ROWS * H_DIM * sizeof(float));
  bf16*  hr_b   = (bf16*)alloc((size_t)ROWS * H_DIM * sizeof(bf16));
  bf16*  rev_b  = (bf16*)alloc((size_t)T_DIM * T_DIM * B_DIM * H_DIM * sizeof(bf16));
  float* sc_r   = (float*)alloc((size_t)T_DIM * T_DIM * B_DIM * sizeof(float));
  float* fsc    = (float*)alloc((size_t)T_DIM * B_DIM * sizeof(float));
  bf16*  ht_b   = (bf16*)alloc((size_t)ROWS * 4 * H_DIM * sizeof(bf16));
  bf16*  ao_b   = (bf16*)alloc((size_t)ROWS * D_DAY * sizeof(bf16));

  auto cvt = [&](const float* s, bf16* d, size_t n) {
    cvt_f2b_k<<<dim3((unsigned)((n + 255) / 256)), 256, 0, stream>>>(s, d, (int)n);
  };
  cvt(x, xbf, (size_t)ROWS * D_IN);
  cvt(Wemb, Wemb_b, (size_t)D_DAY * D_IN);
  cvt(Wihf, Wihf_b, (size_t)G3H * D_DAY);
  cvt(Whhf, Whhf_b, (size_t)G3H * H_DIM);
  cvt(Wihr, Wihr_b, (size_t)G3H * D_DAY);
  cvt(Whhr, Whhr_b, (size_t)G3H * H_DIM);
  cvt(Wao, Wao_b, (size_t)D_DAY * 4 * H_DIM);
  cvt(Wo, Wo_b, (size_t)D_OUTN * D_DAY);
  // pad rows of W_o (942..1023): contents only feed discarded output columns,
  // but keep them finite/deterministic.
  zero_b_k<<<dim3(((D_OUTP - D_OUTN) * D_DAY + 255) / 256), 256, 0, stream>>>(
      Wo_b + (size_t)D_OUTN * D_DAY, (D_OUTP - D_OUTN) * D_DAY);

  zero_f_k<<<dim3((B_DIM * H_DIM + 255) / 256), 256, 0, stream>>>(hf, B_DIM * H_DIM);
  zero_b_k<<<dim3((B_DIM * H_DIM + 255) / 256), 256, 0, stream>>>(hf_b, B_DIM * H_DIM);
  zero_f_k<<<dim3((ROWS * H_DIM + 255) / 256), 256, 0, stream>>>(hr, ROWS * H_DIM);
  zero_b_k<<<dim3((ROWS * H_DIM + 255) / 256), 256, 0, stream>>>(hr_b, ROWS * H_DIM);

  // day_emb (bf16) = x @ W_emb^T + b_emb           (2048 x 256, K=4096)
  {
    dim3 g((D_DAY + 255) / 256, ROWS / 64);
    gemm_bf16_k<0, false, true><<<g, 256, 0, stream>>>(
        xbf, Wemb_b, bemb, nullptr, demb_b, ROWS, D_DAY, D_IN);
  }
  // gi_f / gi_r = day_emb @ Wih^T + bih            (2048 x 768, K=256)
  {
    dim3 g((G3H + 255) / 256, ROWS / 64);
    gemm_bf16_k<0, true, false><<<g, 256, 0, stream>>>(
        demb_b, Wihf_b, bihf, gi_f, nullptr, ROWS, G3H, D_DAY);
    gemm_bf16_k<0, true, false><<<g, 256, 0, stream>>>(
        demb_b, Wihr_b, bihr, gi_r, nullptr, ROWS, G3H, D_DAY);
  }
  // 64 sequential fused recurrent steps: rev matrix (128 blocks) + fwd (2)
  for (int step = 0; step < T_DIM; ++step) {
    gru_step_k<<<dim3(REV_BLOCKS + B_DIM / 16), 32 * STEP_WAVES, 0, stream>>>(
        Whhr_b, bhhr, gi_r, hr, hr_b, rev_b, sc_r, attnw,
        Whhf_b, bhhf, gi_f, hf, hf_b, fwd_o, step);
  }
  // forward attention scores fwd . w_f
  fwd_score_k<<<dim3(ROWS / 8), 256, 0, stream>>>(fwd_o, attnw, fsc);
  // masked softmax + contexts + concat -> h_t (bf16, 2048 x 1024)
  attn_combine_k<<<dim3(ROWS), 256, 0, stream>>>(fsc, sc_r, attnb, fwd_o, rev_b, ht_b);
  // h_t_out (bf16) = h_t @ W_ao^T + b_ao           (2048 x 256, K=1024)
  {
    dim3 g((D_DAY + 255) / 256, ROWS / 64);
    gemm_bf16_k<0, false, true><<<g, 256, 0, stream>>>(
        ht_b, Wao_b, bao, nullptr, ao_b, ROWS, D_DAY, 4 * H_DIM);
  }
  // out = sigmoid(h_t_out @ W_o^T + b_o)           (2048 x 942, K=256)
  {
    dim3 g((D_OUTN + 255) / 256, ROWS / 64);
    gemm_bf16_k<1, true, false><<<g, 256, 0, stream>>>(
        ao_b, Wo_b, bo, out, nullptr, ROWS, D_OUTN, D_DAY);
  }
}
